// Outer_14173392076867
// MI455X (gfx1250) — matrix operand order repaired and checked by reading, not compiled, not run on hardware
//
#include <hip/hip_runtime.h>
#include <hip/hip_bf16.h>

// B=256, H=8, P=64, T=3, D=64
typedef __attribute__((ext_vector_type(16))) __bf16        v16bf;
typedef __attribute__((ext_vector_type(8)))  float         v8f;
typedef __attribute__((ext_vector_type(8)))  unsigned int  v8u;

#define WMMA_BF16(a, b, c)                                                   \
  __builtin_amdgcn_wmma_f32_16x16x32_bf16(false, (a), false, (b), (short)0,  \
                                          (c), false, false)

__device__ __forceinline__ unsigned short f2bf(float f) {
  unsigned u = __builtin_bit_cast(unsigned, f);
  u += 0x7FFFu + ((u >> 16) & 1u);  // round-to-nearest-even
  return (unsigned short)(u >> 16);
}

// ---------------------------------------------------------------------------
// Kernel 0: fc1_w (f32 [64][32768]) -> bf16 in workspace (same layout)
// ---------------------------------------------------------------------------
__global__ __launch_bounds__(256) void cvt_bf16_kernel(
    const float* __restrict__ src, unsigned short* __restrict__ dst, int n) {
  int i = blockIdx.x * 256 + threadIdx.x;
  int stride = gridDim.x * 256;
  for (; i < n; i += stride) dst[i] = f2bf(src[i]);
}

// ---------------------------------------------------------------------------
// Kernel 1: per (n,h): dX = dir*exp(-dist); Xl = dX@Wl, Xr = dX@Wr (WMMA),
// att[i,d] = sum_t Xl*Xr, stored bf16 as flat [256][32768].
// One block (8 waves) per (n,h). ~41.7 KB LDS.
// ---------------------------------------------------------------------------
__global__ __launch_bounds__(256) void pair_att_kernel(
    const float* __restrict__ X, const float* __restrict__ Wl,
    const float* __restrict__ Wr, unsigned short* __restrict__ att) {
  __shared__ float xs[3][64];                               // [t][p]
  __shared__ __align__(16) unsigned short wls[64][64];      // [d][j] bf16
  __shared__ __align__(16) unsigned short wrs[64][64];      // [d][j] bf16
  __shared__ __align__(16) unsigned short dxs[3][64][64];   // [t][i][j] bf16

  const int nb  = blockIdx.x >> 3;   // batch index
  const int h   = blockIdx.x & 7;
  const int tid = threadIdx.x;

  // --- stage X[nb][h][:][:] (64x3 f32) ---
  const float* xg = X + (size_t)(nb * 8 + h) * 64 * 3;
  if (tid < 192) {
    int p = tid / 3, t = tid % 3;
    xs[t][p] = xg[tid];
  }

  // --- stage W_left/W_right[h], transposed to [d][j], bf16 ---
  const float* wlg = Wl + (size_t)h * 4096;
  const float* wrg = Wr + (size_t)h * 4096;
  for (int idx = tid; idx < 4096; idx += 256) {
    int j = idx >> 6, d = idx & 63;
    wls[d][j] = f2bf(wlg[idx]);
    wrs[d][j] = f2bf(wrg[idx]);
  }
  __syncthreads();

  // --- pairwise dX[t][i][j]: 2 transcendentals per element ---
  for (int idx = tid; idx < 3 * 64 * 64; idx += 256) {
    int t = idx >> 12;
    int rem = idx & 4095;
    int i = rem >> 6, j = rem & 63;
    float delta = xs[t][i] - xs[t][j];
    float d2 = delta * delta + 1e-5f;
    float r  = __frsqrt_rn(d2);       // 1/dist
    float dist = d2 * r;              // sqrt(d2)
    float val = delta * r * __expf(-dist);
    dxs[t][i][j] = f2bf(val);
  }
  __syncthreads();

  // --- WMMA stage: wave owns 2 of the 16 (mi,ni) output tiles ---
  const int lane  = tid & 31;
  const int wave  = tid >> 5;
  const int m16   = lane & 15;
  const int abase = (lane < 16) ? 0 : 8;   // C-row offset
  const int klo   = (lane < 16) ? 0 : 16;  // B-frag K offset

  unsigned short* outb = att + (size_t)nb * 32768 + (size_t)h * 4096;

  for (int s = 0; s < 2; ++s) {
    const int tile = wave * 2 + s;
    const int mi = tile >> 2, ni = tile & 3;
    const int i = mi * 16 + m16;   // A row for this lane
    const int d = ni * 16 + m16;   // B column (= output col) for this lane

    const unsigned int* blrow = (const unsigned int*)&wls[d][0];
    const unsigned int* brrow = (const unsigned int*)&wrs[d][0];

    v8f acc = {};
    for (int t = 0; t < 3; ++t) {
      const unsigned int* arow = (const unsigned int*)&dxs[t][i][0];

      v8u a0u, a1u, bl0u, bl1u, br0u, br1u;
      {
        // A 16x32 bf16 lane layout: K = jc + klo + e, contiguous in LDS row
        int o0 = (0 + klo) >> 1, o1 = (8 + klo) >> 1;
        int o2 = (32 + klo) >> 1, o3 = (40 + klo) >> 1;
#pragma unroll
        for (int q = 0; q < 4; ++q) {
          a0u[q] = arow[o0 + q]; a0u[4 + q] = arow[o1 + q];
          a1u[q] = arow[o2 + q]; a1u[4 + q] = arow[o3 + q];
        }
      }
      {
        // B 32x16 bf16 lane layout: K = jc + klo + e, contiguous in LDS row
        int ob0 = (0 + klo) >> 1, ob1 = (32 + klo) >> 1;
#pragma unroll
        for (int q = 0; q < 8; ++q) {
          bl0u[q] = blrow[ob0 + q]; bl1u[q] = blrow[ob1 + q];
          br0u[q] = brrow[ob0 + q]; br1u[q] = brrow[ob1 + q];
        }
      }
      v16bf a0  = __builtin_bit_cast(v16bf, a0u);
      v16bf a1  = __builtin_bit_cast(v16bf, a1u);
      v16bf bl0 = __builtin_bit_cast(v16bf, bl0u);
      v16bf bl1 = __builtin_bit_cast(v16bf, bl1u);
      v16bf br0 = __builtin_bit_cast(v16bf, br0u);
      v16bf br1 = __builtin_bit_cast(v16bf, br1u);

      v8f cl = {};
      cl = WMMA_BF16(a0, bl0, cl);
      cl = WMMA_BF16(a1, bl1, cl);
      v8f cr = {};
      cr = WMMA_BF16(a0, br0, cr);
      cr = WMMA_BF16(a1, br1, cr);
      acc += cl * cr;  // elementwise, reduce over t
    }

    // C/D layout: VGPR r -> M = r + abase, N = m16
#pragma unroll
    for (int r = 0; r < 8; ++r) {
      int m = mi * 16 + r + abase;
      outb[m * 64 + d] = f2bf(acc[r]);
    }
  }
}

// ---------------------------------------------------------------------------
// Kernel 2: split-K fc1 partial GEMM.
// grid (16 mtiles, 16 kchunks) x 128 threads; wave w owns hidden cols 16w..;
// each wave: 64 bf16 WMMAs over its K=2048 slice; unique f32 stores to
// partial[kchunk][256][64] (deterministic, no atomics).
// ---------------------------------------------------------------------------
__global__ __launch_bounds__(128) void fc1_partial_kernel(
    const unsigned short* __restrict__ flat,  // [256][32768] bf16
    const unsigned short* __restrict__ w1,    // [64][32768] bf16
    float* __restrict__ partial) {            // [16][256][64] f32
  const int tid  = threadIdx.x;
  const int wave = tid >> 5, lane = tid & 31;

  const int m16   = lane & 15;
  const int abase = (lane < 16) ? 0 : 8;
  const int klo   = (lane < 16) ? 0 : 16;
  const int mtile  = blockIdx.x;  // 0..15
  const int kchunk = blockIdx.y;  // 0..15
  const int m = mtile * 16 + m16;  // batch row this lane serves for A
  const int n = wave * 16 + m16;   // hidden unit this lane serves for B
  const int kbase = kchunk * 2048;

  const unsigned int* arow = (const unsigned int*)(flat + (size_t)m * 32768);
  const unsigned int* brow = (const unsigned int*)(w1 + (size_t)n * 32768);

  v8f c = {};
#pragma unroll 4
  for (int k0 = kbase; k0 < kbase + 2048; k0 += 32) {
    v8u au, bu;
    int o0 = (k0 + klo) >> 1, o1 = (k0 + 8 + klo) >> 1;
    int ob = (k0 + klo) >> 1;
#pragma unroll
    for (int q = 0; q < 4; ++q) {
      au[q] = arow[o0 + q];
      au[4 + q] = arow[o1 + q];
    }
#pragma unroll
    for (int q = 0; q < 8; ++q) bu[q] = brow[ob + q];
    c = WMMA_BF16(__builtin_bit_cast(v16bf, au),
                  __builtin_bit_cast(v16bf, bu), c);
  }

  float* pout = partial + (size_t)kchunk * 256 * 64;
#pragma unroll
  for (int r = 0; r < 8; ++r) {
    int mr = mtile * 16 + r + abase;
    pout[mr * 64 + n] = c[r];
  }
}

// ---------------------------------------------------------------------------
// Kernel 3: epilogue. out[b] = sum_d silu(sum_kc partial[kc][b][d] + b1[d])
//                              * w2[d]  + b2.   grid 256 x 64 threads.
// ---------------------------------------------------------------------------
__global__ __launch_bounds__(64) void mlp_tail_kernel(
    const float* __restrict__ partial,  // [16][256][64]
    const float* __restrict__ b1, const float* __restrict__ w2,
    const float* __restrict__ b2, float* __restrict__ out) {
  __shared__ float tmp[64];
  const int b = blockIdx.x;
  const int d = threadIdx.x;

  float s = 0.f;
#pragma unroll
  for (int kc = 0; kc < 16; ++kc)
    s += partial[((size_t)kc * 256 + b) * 64 + d];

  float v = s + b1[d];
  tmp[d] = (v / (1.f + __expf(-v))) * w2[d];  // silu * fc2 weight
  __syncthreads();

#pragma unroll
  for (int off = 32; off >= 1; off >>= 1) {
    if (d < off) tmp[d] += tmp[d + off];
    __syncthreads();
  }
  if (d == 0) out[b] = tmp[0] + b2[0];
}

// ---------------------------------------------------------------------------
extern "C" void kernel_launch(void* const* d_in, const int* in_sizes, int n_in,
                              void* d_out, int out_size, void* d_ws,
                              size_t ws_size, hipStream_t stream) {
  const float* X   = (const float*)d_in[0];
  const float* Wl  = (const float*)d_in[1];
  const float* Wr  = (const float*)d_in[2];
  const float* f1w = (const float*)d_in[3];
  const float* f1b = (const float*)d_in[4];
  const float* f2w = (const float*)d_in[5];
  const float* f2b = (const float*)d_in[6];
  float* out = (float*)d_out;

  // workspace: [0,4MB) fc1_w bf16 ; [4MB,20MB) att bf16 ; [20MB,21MB) partial
  unsigned short* w1bf  = (unsigned short*)d_ws;
  unsigned short* attbf = (unsigned short*)d_ws + (size_t)64 * 32768;
  float* partial = (float*)((char*)d_ws + (size_t)(64 + 256) * 32768 * 2);

  cvt_bf16_kernel<<<2048, 256, 0, stream>>>(f1w, w1bf, 64 * 32768);
  pair_att_kernel<<<256 * 8, 256, 0, stream>>>(X, Wl, Wr, attbf);
  fc1_partial_kernel<<<dim3(16, 16), 128, 0, stream>>>(attbf, w1bf, partial);
  mlp_tail_kernel<<<256, 64, 0, stream>>>(partial, f1b, f2w, f2b, out);
}
